// PointSimilarity2_5463198401195
// MI455X (gfx1250) — matrix operand order repaired and checked by reading, compiled, not run on hardware
//
#include <hip/hip_runtime.h>
#include <stdint.h>

#define NPTS 512
#define CIN  128
#define NB   2
#define DROP 52            // 512 - int(512 * 0.9) smallest entries get masked out
#define BN_EPS 1e-5f
#define SLOPE  0.01f
#define NS_OFF ((size_t)NB * NPTS * NPTS)   // node_similarity offset in d_out
#define APAD 20            // row stride (u32) of A-fragment arrays -> conflict-free reads

typedef __attribute__((ext_vector_type(16))) __bf16 v16bf;
typedef __attribute__((ext_vector_type(8)))  float  v8f;

union Frag { uint32_t u[8]; v16bf v; };

__device__ __forceinline__ float hi_f(float x) {
  return __uint_as_float(__float_as_uint(x) & 0xffff0000u);
}
// a -> bf16 in [15:0] (even K), b -> bf16 in [31:16] (odd K); truncating split,
// the residual is captured by the lo fragment (hi+lo carries ~16 mantissa bits).
__device__ __forceinline__ uint32_t pack_bf2(float a, float b) {
  return (__float_as_uint(b) & 0xffff0000u) | (__float_as_uint(a) >> 16);
}

__device__ __forceinline__ float block_sum(float v, float* red, int t) {
  red[t] = v; __syncthreads();
  for (int s = 128; s > 0; s >>= 1) { if (t < s) red[t] += red[t + s]; __syncthreads(); }
  float r = red[0]; __syncthreads();
  return r;
}

__global__ void __launch_bounds__(256)
ps2_fused(const float* __restrict__ vp, const float* __restrict__ ep,
          const float* __restrict__ W1, const float* __restrict__ g1,
          const float* __restrict__ be1, const float* __restrict__ m1,
          const float* __restrict__ v1, const float* __restrict__ W2,
          const float* __restrict__ g2, const float* __restrict__ be2,
          const float* __restrict__ m2, const float* __restrict__ v2,
          const float* __restrict__ W3, const float* __restrict__ b3,
          float* __restrict__ out)
{
  extern __shared__ uint32_t smem[];
  uint32_t* sAh = smem;                       // [64 p][APAD] packed bf16-hi A frags
  uint32_t* sAl = smem + 64 * APAD;           // lo residuals
  float* sVi = (float*)(smem + 2 * 64 * APAD);// vp[b,i,:]      128
  float* sW3 = sVi + 128;                     //                 64
  float* sH1 = sW3 + 64;                      // h1 tile [16][132]
  float* sH2 = sH1 + 16 * 132;                // h2 tile [16][68]
  float* sLg = sH2 + 16 * 68;                 // logit partials [16][16]
  float* sE  = sLg + 256;                     // e row [512]
  float* sSo = sE + 512;                      // sort buffer [512]
  float* sRd = sSo + 512;                     // reduction [256]

  const int t = threadIdx.x;
  const int b = (int)blockIdx.x >> 9;
  const int i = (int)blockIdx.x & (NPTS - 1);
  const int w = t >> 5;                // wave id 0..7 (owns one 16-wide o-tile)
  const int l = t & 15;                // lane-in-half
  const int h = (t >> 4) & 1;          // half of wave32
  const int pb = t & 63;               // channel-pair owned in cooperative builds
  const int rb = t >> 6;               // base row in cooperative builds

  const float* eprow = ep + ((size_t)(b * NPTS + i)) * NPTS;
  float* outep = out + ((size_t)(b * NPTS + i)) * NPTS;
  float* outns = out + NS_OFF + ((size_t)(b * NPTS + i)) * NPTS;
  const float* vpb = vp + (size_t)b * NPTS * CIN;

  // ------------- prolog: vp_i, W3 to LDS -------------
  if (t < 128) sVi[t] = vpb[(size_t)i * CIN + t];
  if (t < 64)  sW3[t] = W3[t];

  // ------------- per-lane B fragments in registers (BN folded into weights) ---
  // GEMM1: wave w owns output cols o1 = 16w..16w+15 (col = lane-in-half).
  Frag B1h[4], B1l[4], B2h[4], B2l[4];
  float bias1, bias2;
  {
    const int o1 = w * 16 + l;
    const float s = g1[o1] * __frsqrt_rn(v1[o1] + BN_EPS);
    bias1 = be1[o1] - m1[o1] * s;
    const float* wr = W1 + o1 * CIN;
    #pragma unroll
    for (int kc = 0; kc < 4; ++kc) {
      #pragma unroll
      for (int v = 0; v < 8; ++v) {
        const int p = kc * 16 + h * 8 + v;          // B K-packing: K = 16h + 2v + s
        float2 g = *(const float2*)(wr + 2 * p);
        float a = g.x * s, c = g.y * s;
        B1h[kc].u[v] = pack_bf2(a, c);
        B1l[kc].u[v] = pack_bf2(a - hi_f(a), c - hi_f(c));
      }
    }
  }
  {
    const int o2 = (w & 3) * 16 + l;   // waves 4..7 build a copy; only 0..3 use it
    const float s = g2[o2] * __frsqrt_rn(v2[o2] + BN_EPS);
    bias2 = be2[o2] - m2[o2] * s;
    const float* wr = W2 + o2 * CIN;
    #pragma unroll
    for (int kc = 0; kc < 4; ++kc) {
      #pragma unroll
      for (int v = 0; v < 8; ++v) {
        const int p = kc * 16 + h * 8 + v;
        float2 g = *(const float2*)(wr + 2 * p);
        float a = g.x * s, c = g.y * s;
        B2h[kc].u[v] = pack_bf2(a, c);
        B2l[kc].u[v] = pack_bf2(a - hi_f(a), c - hi_f(c));
      }
    }
  }
  __syncthreads();

  // ------------- node_similarity_l2 = -||v_i - v_j||^2 (VALU; warms caches) ---
  for (int j = t; j < NPTS; j += 256) {
    const float4* rowp = (const float4*)(vpb + (size_t)j * CIN);
    float acc = 0.f;
    #pragma unroll
    for (int c4 = 0; c4 < CIN / 4; ++c4) {
      float4 g = rowp[c4];
      float d0 = sVi[4*c4+0] - g.x, d1 = sVi[4*c4+1] - g.y;
      float d2 = sVi[4*c4+2] - g.z, d3 = sVi[4*c4+3] - g.w;
      acc += d0*d0 + d1*d1 + d2*d2 + d3*d3;
    }
    outns[j] = -acc;
  }
  __syncthreads();

  // ------------- main loop over 16-wide j tiles -------------
  for (int jt = 0; jt < NPTS / 16; ++jt) {
    const int j0 = jt * 16;

    // ---- cooperative A1 build: diff tile, split+packed ONCE for all 8 waves.
    // A K-pair layout: lane(l,h) chunk kc reads p = 16kc + 8h + v.
    #pragma unroll
    for (int k = 0; k < 4; ++k) {
      const int r = rb + 4 * k;
      float2 g = *(const float2*)(vpb + (size_t)(j0 + r) * CIN + 2 * pb);
      float d0 = sVi[2*pb]     - g.x; d0 *= d0;
      float d1 = sVi[2*pb + 1] - g.y; d1 *= d1;
      sAh[pb * APAD + r] = pack_bf2(d0, d1);
      sAl[pb * APAD + r] = pack_bf2(d0 - hi_f(d0), d1 - hi_f(d1));
      if (jt + 1 < NPTS / 16)
        __builtin_prefetch(vpb + (size_t)(j0 + 16 + r) * CIN + 2 * pb, 0, 3);
    }
    __syncthreads();

    // ---- GEMM1: h1 = leaky(diff @ W1' + b1); 12 WMMAs/wave, B resident in VGPRs
    {
      v8f acc = {0.f,0.f,0.f,0.f,0.f,0.f,0.f,0.f};
      #pragma unroll
      for (int kc = 0; kc < 4; ++kc) {
        Frag Ah, Al;
        #pragma unroll
        for (int v = 0; v < 8; ++v) {
          const int p = kc * 16 + h * 8 + v;
          Ah.u[v] = sAh[p * APAD + l];
          Al.u[v] = sAl[p * APAD + l];
        }
        acc = __builtin_amdgcn_wmma_f32_16x16x32_bf16(false, Ah.v, false, B1h[kc].v, (short)0, acc, false, false);
        acc = __builtin_amdgcn_wmma_f32_16x16x32_bf16(false, Ah.v, false, B1l[kc].v, (short)0, acc, false, false);
        acc = __builtin_amdgcn_wmma_f32_16x16x32_bf16(false, Al.v, false, B1h[kc].v, (short)0, acc, false, false);
      }
      #pragma unroll
      for (int r = 0; r < 8; ++r) {
        float x = acc[r] + bias1;
        x = (x > 0.f) ? x : SLOPE * x;
        sH1[(r + 8 * h) * 132 + w * 16 + l] = x;  // D: row r (+8 hi half), col = lane
      }
    }
    __syncthreads();

    // ---- cooperative A2 build: pack h1 into the (reused) fragment buffers
    #pragma unroll
    for (int k = 0; k < 4; ++k) {
      const int r = rb + 4 * k;
      float2 g = *(const float2*)(sH1 + r * 132 + 2 * pb);
      sAh[pb * APAD + r] = pack_bf2(g.x, g.y);
      sAl[pb * APAD + r] = pack_bf2(g.x - hi_f(g.x), g.y - hi_f(g.y));
    }
    __syncthreads();

    // ---- GEMM2: h2 = leaky(h1 @ W2' + b2); waves 0..3
    if (w < 4) {
      v8f a2 = {0.f,0.f,0.f,0.f,0.f,0.f,0.f,0.f};
      #pragma unroll
      for (int kc = 0; kc < 4; ++kc) {
        Frag Ah, Al;
        #pragma unroll
        for (int v = 0; v < 8; ++v) {
          const int p = kc * 16 + h * 8 + v;
          Ah.u[v] = sAh[p * APAD + l];
          Al.u[v] = sAl[p * APAD + l];
        }
        a2 = __builtin_amdgcn_wmma_f32_16x16x32_bf16(false, Ah.v, false, B2h[kc].v, (short)0, a2, false, false);
        a2 = __builtin_amdgcn_wmma_f32_16x16x32_bf16(false, Ah.v, false, B2l[kc].v, (short)0, a2, false, false);
        a2 = __builtin_amdgcn_wmma_f32_16x16x32_bf16(false, Al.v, false, B2h[kc].v, (short)0, a2, false, false);
      }
      #pragma unroll
      for (int r = 0; r < 8; ++r) {
        float x = a2[r] + bias2;
        x = (x > 0.f) ? x : SLOPE * x;
        sH2[(r + 8 * h) * 68 + w * 16 + l] = x;
      }
    }
    __syncthreads();

    // ---- logits = h2 . W3 + b3; e = sigmoid(logit) * ep_last (diag zeroed)
    {
      const int jr = t >> 4, q = t & 15;
      float p4 = 0.f;
      #pragma unroll
      for (int mm = 0; mm < 4; ++mm) p4 += sH2[jr * 68 + q * 4 + mm] * sW3[q * 4 + mm];
      sLg[jr * 16 + q] = p4;
    }
    __syncthreads();
    if (t < 16) {
      float lg = b3[0];
      #pragma unroll
      for (int q = 0; q < 16; ++q) lg += sLg[t * 16 + q];
      const int j = j0 + t;
      float e = 1.f / (1.f + __expf(-lg));
      sE[j] = (j == i) ? 0.f : e * eprow[j];
    }
    __syncthreads();
  }

  // ---------------- row epilogue ----------------
  // ep_last_sum (diagonal removed)
  float part = 0.f;
  for (int j = t; j < NPTS; j += 256) if (j != i) part += eprow[j];
  const float eplsum = block_sum(part, sRd, t);

  // rank-DROP threshold via bitonic sort (ascending) of the 512 row values
  sSo[t] = sE[t]; sSo[t + 256] = sE[t + 256];
  __syncthreads();
  for (int k = 2; k <= NPTS; k <<= 1) {
    for (int jj = k >> 1; jj > 0; jj >>= 1) {
      #pragma unroll 1
      for (int ii = t; ii < NPTS; ii += 256) {
        const int ixj = ii ^ jj;
        if (ixj > ii) {
          float a = sSo[ii], c = sSo[ixj];
          const bool up = ((ii & k) == 0);
          if (up ? (a > c) : (a < c)) { sSo[ii] = c; sSo[ixj] = a; }
        }
      }
      __syncthreads();
    }
  }
  const float thr = sSo[DROP];   // keep top-(512-DROP) values: x >= thr

  // L1 norm of kept values
  part = 0.f;
  for (int j = t; j < NPTS; j += 256) { float x = sE[j]; if (x >= thr) part += x; }
  const float lsum = block_sum(part, sRd, t);
  const float scale = eplsum / fmaxf(lsum, 1e-12f);

  // rescale, diagonal + 1e-6, then row-normalize
  part = 0.f;
  for (int j = t; j < NPTS; j += 256) {
    float x = sE[j];
    x = (x >= thr) ? x * scale : 0.f;
    x += ((j == i) ? 1.f : 0.f) + 1e-6f;
    sE[j] = x; part += x;
  }
  const float inv = 1.f / block_sum(part, sRd, t);
  for (int j = t; j < NPTS; j += 256) outep[j] = sE[j] * inv;
}

extern "C" void kernel_launch(void* const* d_in, const int* in_sizes, int n_in,
                              void* d_out, int out_size, void* d_ws, size_t ws_size,
                              hipStream_t stream) {
  (void)in_sizes; (void)n_in; (void)d_ws; (void)ws_size; (void)out_size;
  const float* vp  = (const float*)d_in[0];
  const float* ep  = (const float*)d_in[1];
  const float* W1  = (const float*)d_in[2];
  const float* g1  = (const float*)d_in[3];
  const float* be1 = (const float*)d_in[4];
  const float* m1  = (const float*)d_in[5];
  const float* v1  = (const float*)d_in[6];
  const float* W2  = (const float*)d_in[7];
  const float* g2  = (const float*)d_in[8];
  const float* be2 = (const float*)d_in[9];
  const float* m2  = (const float*)d_in[10];
  const float* v2  = (const float*)d_in[11];
  const float* W3  = (const float*)d_in[12];
  const float* b3  = (const float*)d_in[13];
  float* out = (float*)d_out;
  constexpr size_t smem = 7488u * 4u;   // 29.25 KB dynamic LDS
  ps2_fused<<<dim3(NB * NPTS), dim3(256), smem, stream>>>(
      vp, ep, W1, g1, be1, m1, v1, W2, g2, be2, m2, v2, W3, b3, out);
}
